// GNN_node_Virtualnode_45621142618641
// MI455X (gfx1250) — matrix operand order repaired and checked by reading, not compiled, not run on hardware
//
#include <hip/hip_runtime.h>
#include <hip/hip_bf16.h>
#include <stdint.h>

// ---------------------------------------------------------------------------
// MI455X (gfx1250) implementation of GNN_node_Virtualnode reference.
// bf16 v_wmma_f32_16x16x32_bf16 for all GEMMs; async global->LDS staging
// (GLOBAL_LOAD_ASYNC_TO_LDS_B128, ASYNCcnt) with double buffering; fp32 for
// all reductions/BN/atomics.
// ---------------------------------------------------------------------------

typedef __bf16 bf16;
typedef __attribute__((ext_vector_type(16))) __bf16 v16bf;
typedef __attribute__((ext_vector_type(8)))  float  v8f;

#define FLAG_BIAS  1
#define FLAG_RELU  2
#define FLAG_OUTF  4
#define FLAG_OUTB  8   // bf16 out, row-major [M][N]
#define FLAG_YACC  16
#define FLAG_OUTBT 32  // bf16 out, transposed [N][M] (next GEMM's B^T operand)

// Problem constants (match reference).
#define NB    128          // graphs
#define NPG   256          // nodes per graph
#define EPG   2048         // edges per graph
#define EMB   256
#define NNODE (NB*NPG)     // 32768
#define ORDER 3
#define NLAY  5

// LDS tile geometry: 64 rows x 32 bf16, padded row stride 40 halves (80 B,
// 16B-aligned rows; 20-bank row step -> conflict-free fragment loads).
#define TROWS 64
#define TK    32
#define TSTR  40                    // halves
#define TILE_HALVES (TROWS * TSTR)  // 2560
#define TILE_BYTES  (TILE_HALVES * 2)

// ---------------------------------------------------------------------------
// CDNA5 async global->LDS copy (16 B per lane) + ASYNCcnt waits.
// each lane copies 16 B from its global address to its LDS address  (ISA 08_async_tensor §4.4, opcode 98).
// ---------------------------------------------------------------------------
__device__ __forceinline__ void async_copy16(uint32_t lds_addr, const void* gptr) {
  asm volatile("global_load_async_to_lds_b128 %0, %1, off"
               :: "v"(lds_addr), "v"(gptr) : "memory");
}
__device__ __forceinline__ void wait_async4() {
  asm volatile("s_wait_asynccnt 4" ::: "memory");
}
__device__ __forceinline__ void wait_async0() {
  asm volatile("s_wait_asynccnt 0" ::: "memory");
}

// Stage a 64x32 bf16 tile (leading dim ldK elements) into LDS at lds_base.
// 256 16-byte chunks / 128 threads = 2 async issues per thread.
__device__ __forceinline__ void stage_tile(const bf16* __restrict__ g, int ldK,
                                           uint32_t lds_base, int t) {
  const char* gc = (const char*)g;
#pragma unroll
  for (int j = 0; j < 2; ++j) {
    int i = t + j * 128;
    int r = i >> 2, c = i & 3;
    async_copy16(lds_base + (uint32_t)(r * 80 + c * 16),
                 gc + (size_t)r * ((size_t)ldK * 2) + (size_t)c * 16);
  }
}

// ---------------------------------------------------------------------------
// Fragment loads from LDS, CDNA5 ISA 7.12.2 16-bit layouts.
// A (16x32, MxK): lane<16: K=0..7 (e0..7), K=16..23 (e8..15); lane>=16: +8.
// B (32x16, KxN): lane = N; same K order as A: K=0..7, 16..23; lane>=16: +8.
// Both tiles stored K-contiguous (B side is pre-transposed in memory).
// ---------------------------------------------------------------------------
__device__ __forceinline__ v16bf load_fragA(const bf16* tile, int baseRow, int lane) {
  const bf16* p = tile + (baseRow + (lane & 15)) * TSTR + ((lane >> 4) << 3);
  v16bf f;
#pragma unroll
  for (int e = 0; e < 8; ++e) f[e] = p[e];
#pragma unroll
  for (int e = 0; e < 8; ++e) f[8 + e] = p[16 + e];
  return f;
}

__device__ __forceinline__ v16bf load_fragB(const bf16* tile, int baseRow, int lane) {
  const bf16* p = tile + (baseRow + (lane & 15)) * TSTR + ((lane >> 4) << 3);
  v16bf f;
#pragma unroll
  for (int e = 0; e < 16; ++e) f[e] = p[(e < 8) ? e : (e + 8)];
  return f;
}

// ---------------------------------------------------------------------------
// Batched GEMM: C[M,N] = A[M,K](bf16,row-major) * B[K,N], with B supplied
// TRANSPOSED: BT[N,K] row-major. 128 threads = 4 waves; 64x64 block tile;
// each wave a 32x32 subtile (2x2 WMMA frags). Double-buffered async staging.
// ---------------------------------------------------------------------------
__global__ __launch_bounds__(128)
void gemm_bf16_wmma(const bf16* __restrict__ A, const bf16* __restrict__ BT,
                    const float* __restrict__ bias,
                    float* __restrict__ outF, bf16* __restrict__ outB,
                    float* __restrict__ yAcc,
                    int M, int N, int K,
                    long long strideA, long long strideB, long long strideC,
                    int flags)
{
  __shared__ __align__(16) bf16 sm[4 * TILE_HALVES];  // {A,B} x 2 buffers, 20 KB

  const int t    = threadIdx.x;
  const int lane = t & 31;
  const int wave = t >> 5;
  const int wm   = wave >> 1, wn = wave & 1;
  const int bm   = blockIdx.y * 64;
  const int bn   = blockIdx.x * 64;
  const long long zb = blockIdx.z;
  A  += zb * strideA;
  BT += zb * strideB;

  const uint32_t ldsBase = (uint32_t)(uintptr_t)&sm[0];
  const bf16* Arow = A  + (size_t)bm * K;
  const bf16* Brow = BT + (size_t)bn * K;

  v8f c00 = {}, c01 = {}, c10 = {}, c11 = {};

  const int nk = K >> 5;
  // prologue: stage tile 0 into buffer 0
  stage_tile(Arow, K, ldsBase, t);
  stage_tile(Brow, K, ldsBase + TILE_BYTES, t);

  for (int it = 0; it < nk; ++it) {
    const int cur = it & 1;
    const bf16* As = sm + cur * (2 * TILE_HALVES);
    const bf16* Bs = As + TILE_HALVES;

    if (it + 1 < nk) {            // stage next tile into the other buffer
      const int nxt = (it + 1) & 1;
      uint32_t nb = ldsBase + (uint32_t)(nxt * 2 * TILE_BYTES);
      stage_tile(Arow + (size_t)(it + 1) * TK, K, nb, t);
      stage_tile(Brow + (size_t)(it + 1) * TK, K, nb + TILE_BYTES, t);
      wait_async4();              // current tile's 4 copies retired (in-order)
    } else {
      wait_async0();
    }
    __syncthreads();              // tile visible to all waves

    v16bf a0 = load_fragA(As, wm * 32,      lane);
    v16bf a1 = load_fragA(As, wm * 32 + 16, lane);
    v16bf b0 = load_fragB(Bs, wn * 32,      lane);
    v16bf b1 = load_fragB(Bs, wn * 32 + 16, lane);

    c00 = __builtin_amdgcn_wmma_f32_16x16x32_bf16(false, a0, false, b0, (short)0, c00, false, false);
    c01 = __builtin_amdgcn_wmma_f32_16x16x32_bf16(false, a0, false, b1, (short)0, c01, false, false);
    c10 = __builtin_amdgcn_wmma_f32_16x16x32_bf16(false, a1, false, b0, (short)0, c10, false, false);
    c11 = __builtin_amdgcn_wmma_f32_16x16x32_bf16(false, a1, false, b1, (short)0, c11, false, false);
    __syncthreads();              // all reads done before this buffer restages
  }

  // ---- epilogue (C layout: VGPR v -> M = v + 8*(lane>>4), N = lane&15) ----
  const int mloc = (lane >> 4) * 8;
  const int nloc = lane & 15;
  v8f cf[2][2] = {{c00, c01}, {c10, c11}};
#pragma unroll
  for (int fm = 0; fm < 2; ++fm) {
#pragma unroll
    for (int fn = 0; fn < 2; ++fn) {
      v8f c = cf[fm][fn];
      int col = bn + wn * 32 + fn * 16 + nloc;
      float bv = (flags & FLAG_BIAS) ? bias[col] : 0.0f;
#pragma unroll
      for (int v = 0; v < 8; ++v) {
        int row = bm + wm * 32 + fm * 16 + mloc + v;
        long long idx = zb * strideC + (long long)row * N + col;
        float x = c[v] + bv;
        if (flags & FLAG_RELU)  x = fmaxf(x, 0.0f);
        if (flags & FLAG_YACC)  yAcc[idx] += x;
        if (flags & FLAG_OUTF)  outF[idx] = x;
        if (flags & FLAG_OUTB)  outB[idx] = (bf16)x;
        if (flags & FLAG_OUTBT) outB[zb * strideC + (long long)col * M + row] = (bf16)x;
      }
    }
  }
}

// ---------------------------------------------------------------------------
// Small helper kernels
// ---------------------------------------------------------------------------
// bf16 conversion with per-layer [K,N] -> [N,K] transpose (B^T operands)
__global__ void k_conv_T(const float* __restrict__ s, bf16* __restrict__ d,
                         int layers, int K, int N) {
  size_t tid = (size_t)blockIdx.x * 256 + threadIdx.x;
  size_t per = (size_t)K * N;
  if (tid >= (size_t)layers * per) return;
  int l = (int)(tid / per);
  int rem = (int)(tid % per);
  int k = rem / N, n = rem % N;
  d[(size_t)l * per + (size_t)n * K + k] = (bf16)s[tid];
}

// dotv[v] = edge_lin_b + <bond_emb_h[v], edge_lin_w>, v = 0..15
__global__ void k_dotv(const float* __restrict__ bh, const float* __restrict__ w,
                       const float* __restrict__ b, float* __restrict__ dotv) {
  int t = threadIdx.x;
  if (t < 16) {
    float a = b[0];
#pragma unroll
    for (int k = 0; k < 64; ++k) a += bh[t * 64 + k] * w[k];
    dotv[t] = a;
  }
}

// adj[b][src][dst] = sigmoid(dotv[bond_type])
__global__ void k_scatter_adj(const int* __restrict__ lei, const int* __restrict__ ea,
                              const float* __restrict__ dotv, float* __restrict__ A32) {
  int tid = blockIdx.x * 256 + threadIdx.x;
  if (tid >= NB * EPG) return;
  int b = tid >> 11, e = tid & (EPG - 1);
  const int* base = lei + (size_t)b * 2 * EPG;
  int s = base[e], d = base[EPG + e];
  float x = dotv[ea[(size_t)b * EPG + e]];
  float w = 1.0f / (1.0f + __expf(-x));
  A32[(size_t)b * (NPG * NPG) + (size_t)s * NPG + d] = w;
}

// add identity, A[i][j] *= colsum(i)^-1/2 * rowsum(j)^-1/2, emit bf16
__global__ __launch_bounds__(256)
void k_norm_adj(float* __restrict__ A32, bf16* __restrict__ Abf) {
  __shared__ float cInv[NPG], rInv[NPG];
  int b = blockIdx.x, t = threadIdx.x;
  float* Ab = A32 + (size_t)b * NPG * NPG;
  Ab[t * NPG + t] += 1.0f;                       // adj + I (diag owned by thread t)
  float rs = 0.f, cs = 0.f;
  for (int j = 0; j < NPG; ++j) rs += Ab[t * NPG + j];   // rowsum(row t)
  for (int i = 0; i < NPG; ++i) cs += Ab[i * NPG + t];   // colsum(col t)
  cInv[t] = rsqrtf(cs);
  rInv[t] = rsqrtf(rs);
  __syncthreads();
  bf16* Ob = Abf + (size_t)b * NPG * NPG;
  for (int i = 0; i < NPG; ++i)
    Ob[i * NPG + t] = (bf16)(Ab[i * NPG + t] * cInv[i] * rInv[t]);
}

// fea gather: y[node][emb] = atom_emb[atom_idx]; xT[b][emb][node] = bf16(.)
__global__ void k_gather(const int* __restrict__ aidx, const float* __restrict__ emb,
                         float* __restrict__ y, bf16* __restrict__ xbT) {
  size_t tid = (size_t)blockIdx.x * 256 + threadIdx.x;
  int b = (int)(tid >> 16), c = (int)((tid >> 8) & 255), i = (int)(tid & 255);
  float v = emb[(size_t)aidx[b * NPG + i] * EMB + c];
  y[((size_t)b * NPG + i) * EMB + c] = v;
  xbT[(size_t)b * NPG * EMB + (size_t)c * NPG + i] = (bf16)v;   // coalesced
}

__global__ void k_scale(const float* __restrict__ y, float* __restrict__ h) {
  size_t tid = (size_t)blockIdx.x * 256 + threadIdx.x;
  h[tid] = y[tid] * (1.0f / (ORDER + 1.0f));
}

__global__ void k_vn_init(const float* __restrict__ vn_emb, float* __restrict__ vn) {
  int tid = blockIdx.x * 256 + threadIdx.x;
  vn[tid] = vn_emb[tid & 255];
}

__global__ void k_add_vn(float* __restrict__ h, const float* __restrict__ vn) {
  size_t tid = (size_t)blockIdx.x * 256 + threadIdx.x;
  int g = (int)(tid >> 16), c = (int)(tid & 255);
  h[tid] += vn[g * EMB + c];
}

// per (graph, 64-channel slice): LDS segment-sum of relu(h[src]+eemb) into dst,
// then z = (1+eps)*h + agg (f32 + bf16 outputs)
__global__ __launch_bounds__(256)
void k_scatter_agg(const float* __restrict__ hin, const int* __restrict__ lei,
                   const int* __restrict__ ea, const float* __restrict__ bond_emb_l,
                   const float* __restrict__ gin_eps, int l,
                   float* __restrict__ z, bf16* __restrict__ zb) {
  __shared__ float agg[NPG * 64];                 // 64 KB
  const int t = threadIdx.x;
  const int b = blockIdx.y;
  const int chbase = blockIdx.x * 64;
  const int ch = t & 63;
  const int eo = t >> 6;

  for (int i = t; i < NPG * 64; i += 256) agg[i] = 0.0f;

  float eembv[16];
  const float* bel = bond_emb_l + (size_t)l * 16 * EMB;
#pragma unroll
  for (int v = 0; v < 16; ++v) eembv[v] = bel[v * EMB + chbase + ch];
  const float epsv = 1.0f + gin_eps[l];
  __syncthreads();

  const int* srcp = lei + (size_t)b * 2 * EPG;
  const int* dstp = srcp + EPG;
  const int* eap  = ea + (size_t)b * EPG;
  const float* hb = hin + ((size_t)b * NPG) * EMB + chbase;

  for (int e0 = 0; e0 < EPG; e0 += 4) {
    int e = e0 + eo;
    int s = srcp[e], d = dstp[e], bv = eap[e];
    float m = hb[(size_t)s * EMB + ch] + eembv[bv];
    if (m > 0.0f) atomicAdd(&agg[d * 64 + ch], m);   // ds_add_f32
  }
  __syncthreads();

  for (int i = t; i < NPG * 64; i += 256) {
    int node = i >> 6, c = i & 63;
    size_t gi = ((size_t)b * NPG + node) * EMB + chbase + c;
    float zv = epsv * hin[gi] + agg[i];
    z[gi]  = zv;
    zb[gi] = (bf16)zv;
  }
}

// BN stats: sums[0..C) = sum, sums[C..2C) = sumsq, over 32768 rows
__global__ __launch_bounds__(256)
void k_bn_stats(const float* __restrict__ z, int C, float* __restrict__ sums) {
  int t = threadIdx.x;
  int r0 = blockIdx.x * 128;
  float a0 = 0.f, q0 = 0.f, a1 = 0.f, q1 = 0.f;
  for (int r = r0; r < r0 + 128; ++r) {
    const float* row = z + (size_t)r * C;
    float v = row[t]; a0 += v; q0 += v * v;
    if (C == 512) { float u = row[t + 256]; a1 += u; q1 += u * u; }
  }
  atomicAdd(&sums[t], a0);
  atomicAdd(&sums[C + t], q0);
  if (C == 512) { atomicAdd(&sums[t + 256], a1); atomicAdd(&sums[C + t + 256], q1); }
}

__global__ void k_bn_final(int C, const float* __restrict__ g, const float* __restrict__ bb,
                           const float* __restrict__ sums, float* __restrict__ ss) {
  int t = blockIdx.x * 256 + threadIdx.x;
  if (t >= C) return;
  const float inv = 1.0f / (float)NNODE;
  float mu  = sums[t] * inv;
  float var = sums[C + t] * inv - mu * mu;
  float sc  = g[t] * rsqrtf(var + 1e-5f);
  ss[t]     = sc;
  ss[C + t] = bb[t] - mu * sc;
}

__global__ void k_bn_apply(const float* __restrict__ z, int C, const float* __restrict__ ss,
                           int relu, float* __restrict__ outF, bf16* __restrict__ outB) {
  size_t tid = (size_t)blockIdx.x * 256 + threadIdx.x;
  int c = (int)(tid % (size_t)C);
  float v = z[tid] * ss[c] + ss[C + c];
  if (relu) v = fmaxf(v, 0.0f);
  if (outF) outF[tid] = v;
  if (outB) outB[tid] = (bf16)v;
}

// vt[b] = sum_nodes(h_in) + vn[b], emitted as bf16 (GEMM A operand)
__global__ __launch_bounds__(256)
void k_vt_seg(const float* __restrict__ hin, const float* __restrict__ vn,
              bf16* __restrict__ vtb) {
  int b = blockIdx.x, c = threadIdx.x;
  float acc = vn[b * EMB + c];
  const float* hb = hin + ((size_t)b * NPG) * EMB;
  for (int i = 0; i < NPG; ++i) acc += hb[(size_t)i * EMB + c];
  vtb[b * EMB + c] = (bf16)acc;
}

// ---------------------------------------------------------------------------
// Launcher
// ---------------------------------------------------------------------------
extern "C" void kernel_launch(void* const* d_in, const int* in_sizes, int n_in,
                              void* d_out, int out_size, void* d_ws, size_t ws_size,
                              hipStream_t stream) {
  const int*   atom_idx   = (const int*)d_in[0];
  const int*   lei        = (const int*)d_in[1];
  const int*   ea         = (const int*)d_in[2];
  /* d_in[3] = order (==3, hardcoded: loop trip must be host-known)        */
  const float* atom_emb   = (const float*)d_in[4];
  const float* bond_emb_h = (const float*)d_in[5];
  const float* edge_lin_w = (const float*)d_in[6];
  const float* edge_lin_b = (const float*)d_in[7];
  const float* bond_emb_l = (const float*)d_in[8];
  const float* gin_eps    = (const float*)d_in[9];
  const float* mlp_w1     = (const float*)d_in[10];
  const float* mlp_b1     = (const float*)d_in[11];
  const float* mlp_bn_g   = (const float*)d_in[12];
  const float* mlp_bn_b   = (const float*)d_in[13];
  const float* mlp_w2     = (const float*)d_in[14];
  const float* mlp_b2     = (const float*)d_in[15];
  const float* out_bn_g   = (const float*)d_in[16];
  const float* out_bn_b   = (const float*)d_in[17];
  const float* vn_emb     = (const float*)d_in[18];
  const float* vn_w1      = (const float*)d_in[19];
  const float* vn_b1      = (const float*)d_in[20];
  const float* vn_w2      = (const float*)d_in[21];
  const float* vn_b2      = (const float*)d_in[22];

  // ---- workspace layout (explicit temporal reuse) ----
  char* W = (char*)d_ws;
  const size_t MiB = 1ull << 20;
  float* A32   = (float*)(W + 0);          // 32 MiB, dead after k_norm_adj
  bf16*  xb0T  = (bf16*)(W + 0);           // 16 MiB (reuses A32 low half)
  bf16*  xb1T  = (bf16*)(W + 16 * MiB);    // 16 MiB
  bf16*  Abf   = (bf16*)(W + 32 * MiB);    // 16 MiB
  float* ybuf  = (float*)(W + 48 * MiB);   // 32 MiB
  float* z1    = (float*)(W + 0);          // 64 MiB (GIN phase, P buffers dead)
  float* hbuf  = (float*)(W + 80 * MiB);   // 32 MiB
  float* zbuf  = (float*)(W + 112 * MiB);  // 32 MiB
  bf16*  zbf   = (bf16*)(W + 144 * MiB);   // 16 MiB
  bf16*  a1n   = (bf16*)(W + 160 * MiB);   // 32 MiB
  float* z2    = (float*)(W + 192 * MiB);  // 32 MiB
  char* S = W + 224 * MiB;
  bf16*  w1T   = (bf16*)S; S += (size_t)NLAY * EMB * 512 * 2;        // [l][512][256]
  bf16*  w2T   = (bf16*)S; S += (size_t)NLAY * 512 * EMB * 2;        // [l][256][512]
  bf16*  vnw1T = (bf16*)S; S += (size_t)(NLAY - 1) * EMB * 512 * 2;
  bf16*  vnw2T = (bf16*)S; S += (size_t)(NLAY - 1) * 512 * EMB * 2;
  float* vnbuf = (float*)S; S += (size_t)NB * EMB * 4;
  bf16*  vtb   = (bf16*)S;  S += (size_t)NB * EMB * 2;
  bf16*  ub    = (bf16*)S;  S += (size_t)NB * 512 * 2;
  float* bnsum = (float*)S; S += 4096;   // sum | sumsq (max C=512)
  float* bnss  = (float*)S; S += 4096;   // scale | shift
  float* dotv  = (float*)S; S += 64;

  // ---- weight conversion to transposed bf16 (B^T operands) ----
  {
    int n1 = NLAY * EMB * 512;
    k_conv_T<<<(n1 + 255) / 256, 256, 0, stream>>>(mlp_w1, w1T, NLAY, 256, 512);
    k_conv_T<<<(n1 + 255) / 256, 256, 0, stream>>>(mlp_w2, w2T, NLAY, 512, 256);
    int n2 = (NLAY - 1) * EMB * 512;
    k_conv_T<<<(n2 + 255) / 256, 256, 0, stream>>>(vn_w1, vnw1T, NLAY - 1, 256, 512);
    k_conv_T<<<(n2 + 255) / 256, 256, 0, stream>>>(vn_w2, vnw2T, NLAY - 1, 512, 256);
  }

  // ---- learned-adjacency dense propagation ----
  k_dotv<<<1, 64, 0, stream>>>(bond_emb_h, edge_lin_w, edge_lin_b, dotv);
  hipMemsetAsync(A32, 0, 32 * MiB, stream);
  k_scatter_adj<<<(NB * EPG + 255) / 256, 256, 0, stream>>>(lei, ea, dotv, A32);
  k_norm_adj<<<NB, 256, 0, stream>>>(A32, Abf);
  k_gather<<<NNODE, 256, 0, stream>>>(atom_idx, atom_emb, ybuf, xb0T);

  bf16* xbp[2] = { xb0T, xb1T };
  int cur = 0;
  for (int s = 0; s < ORDER; ++s) {       // x = A@x ; y += x ; write x^T bf16
    gemm_bf16_wmma<<<dim3(4, 4, NB), 128, 0, stream>>>(
        Abf, xbp[cur], nullptr, nullptr, xbp[1 - cur], ybuf,
        NPG, EMB, NPG, (long long)NPG * NPG, (long long)NPG * EMB,
        (long long)NPG * EMB, FLAG_YACC | FLAG_OUTBT);
    cur ^= 1;
  }
  k_scale<<<NNODE, 256, 0, stream>>>(ybuf, hbuf);

  // ---- virtual-node GIN stack ----
  k_vn_init<<<NB, 256, 0, stream>>>(vn_emb, vnbuf);
  for (int l = 0; l < NLAY; ++l) {
    k_add_vn<<<NNODE, 256, 0, stream>>>(hbuf, vnbuf);  // h_in (in place)
    k_scatter_agg<<<dim3(4, NB), 256, 0, stream>>>(hbuf, lei, ea, bond_emb_l,
                                                   gin_eps, l, zbuf, zbf);
    // z @ w1 + b1 -> z1 (f32)
    gemm_bf16_wmma<<<dim3(8, NNODE / 64, 1), 128, 0, stream>>>(
        zbf, w1T + (size_t)l * EMB * 512, mlp_b1 + l * 512,
        z1, nullptr, nullptr, NNODE, 512, EMB, 0, 0, 0, FLAG_BIAS | FLAG_OUTF);
    hipMemsetAsync(bnsum, 0, 4096, stream);
    k_bn_stats<<<NNODE / 128, 256, 0, stream>>>(z1, 512, bnsum);
    k_bn_final<<<2, 256, 0, stream>>>(512, mlp_bn_g + l * 512, mlp_bn_b + l * 512,
                                      bnsum, bnss);
    k_bn_apply<<<(NNODE * 512) / 256, 256, 0, stream>>>(z1, 512, bnss, 1,
                                                        nullptr, a1n);
    // relu(bn(z1)) @ w2 + b2 -> z2 (f32)
    gemm_bf16_wmma<<<dim3(4, NNODE / 64, 1), 128, 0, stream>>>(
        a1n, w2T + (size_t)l * 512 * EMB, mlp_b2 + l * EMB,
        z2, nullptr, nullptr, NNODE, EMB, 512, 0, 0, 0, FLAG_BIAS | FLAG_OUTF);
    hipMemsetAsync(bnsum, 0, 4096, stream);
    k_bn_stats<<<NNODE / 128, 256, 0, stream>>>(z2, 256, bnsum);
    k_bn_final<<<1, 256, 0, stream>>>(256, out_bn_g + l * EMB, out_bn_b + l * EMB,
                                      bnsum, bnss);
    if (l < NLAY - 1) {
      // virtual-node update (consumes h_in before it is overwritten)
      k_vt_seg<<<NB, 256, 0, stream>>>(hbuf, vnbuf, vtb);
      gemm_bf16_wmma<<<dim3(8, 2, 1), 128, 0, stream>>>(
          vtb, vnw1T + (size_t)l * EMB * 512, vn_b1 + l * 512,
          nullptr, ub, nullptr, NB, 512, EMB, 0, 0, 0,
          FLAG_BIAS | FLAG_RELU | FLAG_OUTB);
      gemm_bf16_wmma<<<dim3(4, 2, 1), 128, 0, stream>>>(
          ub, vnw2T + (size_t)l * 512 * EMB, vn_b2 + l * EMB,
          vnbuf, nullptr, nullptr, NB, EMB, 512, 0, 0, 0,
          FLAG_BIAS | FLAG_RELU | FLAG_OUTF);
      // h = relu(bn(z2))
      k_bn_apply<<<(NNODE * EMB) / 256, 256, 0, stream>>>(z2, 256, bnss, 1,
                                                          hbuf, nullptr);
    } else {
      // final layer: out = bn(z2), no relu (JK='last')
      k_bn_apply<<<(NNODE * EMB) / 256, 256, 0, stream>>>(z2, 256, bnss, 0,
                                                          (float*)d_out, nullptr);
    }
  }
  (void)in_sizes; (void)n_in; (void)out_size; (void)ws_size;
}
